// GCNHet_14087492731176
// MI455X (gfx1250) — matrix operand order repaired and checked by reading, not compiled, not run on hardware
//
#include <hip/hip_runtime.h>
#include <hip/hip_bf16.h>

typedef __attribute__((ext_vector_type(16))) _Float16 v16h;
typedef __attribute__((ext_vector_type(8)))  _Float16 v8h;
typedef __attribute__((ext_vector_type(8)))  float    v8f;

#define KMAX 320   // padded K: DIN=300 -> 320, D=128 -> 128

// ---------------------------------------------------------------------------
// One-shot weight prep: WT[n][k] = f16(W[k][n]), zero-padded to Kpad.
// Puts B fragments in a contiguous, branch-free layout for the GEMM hot loop.
// ---------------------------------------------------------------------------
__global__ void gcnhet_wcvt(const float* __restrict__ W,   // [K,128]
                            _Float16* __restrict__ WT,     // [128,Kpad]
                            int K, int Kpad) {
  int i = blockIdx.x * blockDim.x + threadIdx.x;           // over 128*Kpad
  if (i >= 128 * Kpad) return;
  int n = i / Kpad, k = i - n * Kpad;
  WT[i] = (_Float16)((k < K) ? W[(size_t)k * 128 + n] : 0.0f);
}

// ---------------------------------------------------------------------------
// C[M,128] = (A[rows?,K] @ W[K,128]) * rowscale[:,None] + bias[None,:]
// 256-thread block (8 wave32) -> 16x128 tile via v_wmma_f32_16x16x32_f16.
// A tile staged in LDS f16 (zero-padded); B read as two contiguous b128
// loads per k-step from the pre-transposed, pre-padded WT.
// ---------------------------------------------------------------------------
__global__ __launch_bounds__(256)
void gcnhet_gemm_wmma(const float* __restrict__ A,        // lda == K
                      const int*   __restrict__ rowidx,   // null => identity
                      int K,
                      const _Float16* __restrict__ WT,    // [128,Kpad] padded
                      const float* __restrict__ bias,     // [128] or null
                      const float* __restrict__ rowscale, // [M] or null
                      float* __restrict__ out)            // [M,128]
{
  __shared__ __align__(16) _Float16 As[16 * KMAX];
  const int tid  = threadIdx.x;
  const int blk  = blockIdx.x;
  const int Kpad = (K + 31) & ~31;

  // cooperative load + f32->f16 convert of the 16xK A tile (gathered rows ok)
  for (int i = tid; i < 16 * Kpad; i += 256) {
    int r = i / Kpad, k = i - r * Kpad;
    int row  = blk * 16 + r;
    int grow = rowidx ? rowidx[row] : row;
    float v  = (k < K) ? A[(size_t)grow * K + k] : 0.0f;
    As[r * KMAX + k] = (_Float16)v;
  }
  __syncthreads();

  const int wave = tid >> 5;        // N-tile: cols [16*wave, 16*wave+16)
  const int lane = tid & 31;
  const int m    = lane & 15;
  const int hs   = lane >> 4;       // half-select per ISA fragment layout
  const int n    = wave * 16 + m;   // this lane's output column

  const _Float16* __restrict__ wrow = WT + (size_t)n * Kpad;

  v8f acc = {};
  for (int kk = 0; kk < Kpad; kk += 32) {
    // A fragment 16x32 (ISA 7.12.2): lanes<16 K=0-7/16-23, lanes>=16 K=8-15/24-31
    v8h alo = *(const v8h*)&As[m * KMAX + kk + hs * 8];
    v8h ahi = *(const v8h*)&As[m * KMAX + kk + 16 + hs * 8];
    // B fragment 32x16: lanes<16 K=0-7/16-23, lanes>=16 K=8-15/24-31
    v8h blo = *(const v8h*)&wrow[kk + hs * 8];
    v8h bhi = *(const v8h*)&wrow[kk + 16 + hs * 8];
    v16h a, b;
#pragma unroll
    for (int j = 0; j < 8; ++j) {
      a[j] = alo[j]; a[8 + j] = ahi[j];
      b[j] = blo[j]; b[8 + j] = bhi[j];
    }
    acc = __builtin_amdgcn_wmma_f32_16x16x32_f16(
        /*neg_a=*/false, a, /*neg_b=*/false, b,
        /*c_mod=*/(short)0, acc, /*reuse_a=*/false, /*reuse_b=*/false);
  }

  const float bv = bias ? bias[n] : 0.0f;
#pragma unroll
  for (int r = 0; r < 8; ++r) {
    int   mr  = r + hs * 8;               // C layout: VGPR r -> M = r + 8*hs
    int   row = blk * 16 + mr;
    float v   = acc[r];
    if (rowscale) v *= rowscale[row];
    v += bv;
    out[(size_t)row * 128 + n] = v;
  }
}

// ---------------------------------------------------------------------------
__global__ void gcnhet_fill_zero(float* __restrict__ p, long long n) {
  long long i = blockIdx.x * (long long)blockDim.x + threadIdx.x;
  long long s = (long long)gridDim.x * blockDim.x;
  for (; i < n; i += s) p[i] = 0.0f;
}

// acc[dst[e], :] += h[src[e], :] * (ew ? ew[e] : 1)   (32 lanes/edge, 4 f/lane)
__global__ void gcnhet_scatter_add(const float* __restrict__ h,
                                   const int* __restrict__ src,
                                   const int* __restrict__ dst,
                                   const float* __restrict__ ew,
                                   float* __restrict__ acc, int E) {
  long long t = blockIdx.x * (long long)blockDim.x + threadIdx.x;
  int e = (int)(t >> 5), lane = (int)(t & 31);
  if (e >= E) return;
  int s = src[e], d = dst[e];
  float w = ew ? ew[e] : 1.0f;
  float4 v = ((const float4*)(h + (size_t)s * 128))[lane];
  float* p = acc + (size_t)d * 128 + lane * 4;
  atomicAdd(p + 0, v.x * w);
  atomicAdd(p + 1, v.y * w);
  atomicAdd(p + 2, v.z * w);
  atomicAdd(p + 3, v.w * w);
}

__global__ void gcnhet_deg_count(const int* __restrict__ dst,
                                 float* __restrict__ deg, int E) {
  int e = blockIdx.x * blockDim.x + threadIdx.x;
  if (e < E) atomicAdd(&deg[dst[e]], 1.0f);
}

// h = relu(h * norm[row] + bias[col])
__global__ void gcnhet_gcn_finish(float* __restrict__ h,
                                  const float* __restrict__ norm,
                                  const float* __restrict__ bias,
                                  long long total) {
  long long i = blockIdx.x * (long long)blockDim.x + threadIdx.x;
  if (i >= total) return;
  int row = (int)(i >> 7), col = (int)(i & 127);
  h[i] = fmaxf(h[i] * norm[row] + bias[col], 0.0f);
}

// acc /= max(deg[row], 1)
__global__ void gcnhet_mean_div(float* __restrict__ acc,
                                const float* __restrict__ deg,
                                long long total) {
  long long i = blockIdx.x * (long long)blockDim.x + threadIdx.x;
  if (i >= total) return;
  acc[i] /= fmaxf(deg[i >> 7], 1.0f);
}

// doc = relu(awd/max(dwd,1) + atd/max(dtd,1))  (written into awd)
__global__ void gcnhet_doc_combine(float* __restrict__ awd,
                                   const float* __restrict__ atd,
                                   const float* __restrict__ dwd,
                                   const float* __restrict__ dtd,
                                   long long total) {
  long long i = blockIdx.x * (long long)blockDim.x + threadIdx.x;
  if (i >= total) return;
  int row = (int)(i >> 7);
  float v = awd[i] / fmaxf(dwd[row], 1.0f) + atd[i] / fmaxf(dtd[row], 1.0f);
  awd[i] = fmaxf(v, 0.0f);
}

// pooled[g, c] = max over docs (doc >= 0 so uint-bit atomicMax is order-exact)
__global__ void gcnhet_pool_max(const float* __restrict__ doc,
                                const int* __restrict__ gid,
                                unsigned* __restrict__ pooled,
                                long long total) {
  long long i = blockIdx.x * (long long)blockDim.x + threadIdx.x;
  if (i >= total) return;
  int row = (int)(i >> 7), col = (int)(i & 127);
  atomicMax(&pooled[(size_t)gid[row] * 128 + col], __float_as_uint(doc[i]));
}

// z = pooled @ outW + outb; loss = mean BCE-with-logits; y_pred = sigmoid(z)
__global__ void gcnhet_final(const float* __restrict__ pooled, // [16,128]
                             const float* __restrict__ outW,   // [128]
                             const float* __restrict__ outb,   // [1]
                             const float* __restrict__ y,      // [16]
                             float* __restrict__ out) {        // [17]
  __shared__ float zs[16];
  int b = threadIdx.x >> 5, lane = threadIdx.x & 31;
  float s = 0.0f;
  for (int c = lane; c < 128; c += 32) s += pooled[b * 128 + c] * outW[c];
  for (int off = 16; off > 0; off >>= 1) s += __shfl_down(s, off, 32);
  if (lane == 0) zs[b] = s + outb[0];
  __syncthreads();
  if (threadIdx.x == 0) {
    float loss = 0.0f;
    for (int i = 0; i < 16; ++i) {
      float z = zs[i];
      loss += fmaxf(z, 0.0f) - z * y[i] + log1pf(expf(-fabsf(z)));
    }
    out[0] = loss / 16.0f;
  }
  if (lane == 0) out[1 + b] = 1.0f / (1.0f + expf(-zs[b]));
}

// ---------------------------------------------------------------------------
static inline unsigned cdivll(long long a, long long b) {
  return (unsigned)((a + b - 1) / b);
}

extern "C" void kernel_launch(void* const* d_in, const int* in_sizes, int n_in,
                              void* d_out, int out_size, void* d_ws, size_t ws_size,
                              hipStream_t stream) {
  (void)n_in; (void)out_size; (void)ws_size;
  const float* word_embeds = (const float*)d_in[0];
  // d_in[1] topic_embeds, d_in[2] doc_gen_embeds, d_in[4] norm_topic,
  // d_in[9] Wt, d_in[10] bt, d_in[20] topic_id, d_in[23..24] tt edges:
  // provably dead in the forward dataflow — skipped.
  const float* norm_word = (const float*)d_in[3];
  const float* Ww0  = (const float*)d_in[5];
  const float* bw0  = (const float*)d_in[6];
  const float* Ww1  = (const float*)d_in[7];
  const float* bw1  = (const float*)d_in[8];
  const float* Wh   = (const float*)d_in[11];
  const float* bh   = (const float*)d_in[12];
  const float* outW = (const float*)d_in[13];
  const float* outb = (const float*)d_in[14];
  const float* ew_wt = (const float*)d_in[15];
  const float* ew_wd = (const float*)d_in[16];
  const float* ew_td = (const float*)d_in[17];
  const float* y_data = (const float*)d_in[18];
  const int* word_id = (const int*)d_in[19];
  const int* src_ww = (const int*)d_in[21];
  const int* dst_ww = (const int*)d_in[22];
  const int* src_wt = (const int*)d_in[25];
  const int* dst_wt = (const int*)d_in[26];
  const int* src_wd = (const int*)d_in[27];
  const int* dst_wd = (const int*)d_in[28];
  const int* src_td = (const int*)d_in[29];
  const int* dst_td = (const int*)d_in[30];
  const int* doc_gid = (const int*)d_in[31];

  const int D   = 128;
  const int NW  = in_sizes[3];            // 100000
  const int NT  = in_sizes[4];            // 800
  const int ND  = in_sizes[31];           // 4096
  const int DIN = in_sizes[5] / D;        // 300
  const int E_ww = in_sizes[21], E_wt = in_sizes[25];
  const int E_wd = in_sizes[27], E_td = in_sizes[29];

  // workspace layout (~108 MB f32)
  float* ws = (float*)d_ws;
  size_t o = 0;
  float* wA   = ws + o; o += (size_t)NW * D;
  float* wB   = ws + o; o += (size_t)NW * D;
  float* tACC = ws + o; o += (size_t)NT * D;
  float* tMSG = ws + o; o += (size_t)NT * D;
  float* dWD  = ws + o; o += (size_t)ND * D;
  float* dTD  = ws + o; o += (size_t)ND * D;
  float* deg_wt = ws + o; o += NT;
  float* deg_wd = ws + o; o += ND;
  float* deg_td = ws + o; o += ND;
  float* pooled = ws + o; o += 16 * D;
  _Float16* WT = (_Float16*)(ws + o); o += (size_t)D * KMAX / 2;  // f16 weight

  const int TB = 256;
  auto ZERO = [&](float* p, long long n) {
    gcnhet_fill_zero<<<cdivll(n, 1024), 1024, 0, stream>>>(p, n);
  };
  auto WCVT = [&](const float* W, int K) {
    int Kpad = (K + 31) & ~31;
    gcnhet_wcvt<<<cdivll((long long)D * Kpad, TB), TB, 0, stream>>>(W, WT, K, Kpad);
  };
  const size_t DD = (size_t)D * D;

  // ---- word GCN layer 1: h = relu(segsum((E[id]@Ww0)*nw)[ww] * nw + bw0)
  WCVT(Ww0, DIN);
  gcnhet_gemm_wmma<<<NW / 16, TB, 0, stream>>>(word_embeds, word_id, DIN, WT,
                                               nullptr, norm_word, wA);
  ZERO(wB, (long long)NW * D);
  gcnhet_scatter_add<<<cdivll((long long)E_ww * 32, TB), TB, 0, stream>>>(
      wA, src_ww, dst_ww, nullptr, wB, E_ww);
  gcnhet_gcn_finish<<<cdivll((long long)NW * D, TB), TB, 0, stream>>>(
      wB, norm_word, bw0, (long long)NW * D);

  // ---- word GCN layer 2
  WCVT(Ww1, D);
  gcnhet_gemm_wmma<<<NW / 16, TB, 0, stream>>>(wB, nullptr, D, WT,
                                               nullptr, norm_word, wA);
  ZERO(wB, (long long)NW * D);
  gcnhet_scatter_add<<<cdivll((long long)E_ww * 32, TB), TB, 0, stream>>>(
      wA, src_ww, dst_ww, nullptr, wB, E_ww);
  gcnhet_gcn_finish<<<cdivll((long long)NW * D, TB), TB, 0, stream>>>(
      wB, norm_word, bw1, (long long)NW * D);   // word final -> wB

  // ---- degrees
  ZERO(deg_wt, NT);
  gcnhet_deg_count<<<cdivll(E_wt, TB), TB, 0, stream>>>(dst_wt, deg_wt, E_wt);
  ZERO(deg_wd, ND);
  gcnhet_deg_count<<<cdivll(E_wd, TB), TB, 0, stream>>>(dst_wd, deg_wd, E_wd);
  ZERO(deg_td, ND);
  gcnhet_deg_count<<<cdivll(E_td, TB), TB, 0, stream>>>(dst_td, deg_td, E_td);

  // ---- t_new0 = mean_agg(word @ Wh[0,0] + bh[0,0], ew_wt)   (live topic state)
  WCVT(Wh + 0 * DD, D);
  gcnhet_gemm_wmma<<<NW / 16, TB, 0, stream>>>(wB, nullptr, D, WT,
                                               bh + 0 * D, nullptr, wA);
  ZERO(tACC, (long long)NT * D);
  gcnhet_scatter_add<<<cdivll((long long)E_wt * 32, TB), TB, 0, stream>>>(
      wA, src_wt, dst_wt, ew_wt, tACC, E_wt);
  gcnhet_mean_div<<<cdivll((long long)NT * D, TB), TB, 0, stream>>>(
      tACC, deg_wt, (long long)NT * D);

  // ---- d_new(l=1) wd part: word @ Wh[1,1] + bh[1,1]
  WCVT(Wh + 4 * DD, D);
  gcnhet_gemm_wmma<<<NW / 16, TB, 0, stream>>>(wB, nullptr, D, WT,
                                               bh + 4 * D, nullptr, wA);
  ZERO(dWD, (long long)ND * D);
  gcnhet_scatter_add<<<cdivll((long long)E_wd * 32, TB), TB, 0, stream>>>(
      wA, src_wd, dst_wd, ew_wd, dWD, E_wd);

  // ---- d_new(l=1) td part: t_new0 @ Wh[1,2] + bh[1,2]
  WCVT(Wh + 5 * DD, D);
  gcnhet_gemm_wmma<<<NT / 16, TB, 0, stream>>>(tACC, nullptr, D, WT,
                                               bh + 5 * D, nullptr, tMSG);
  ZERO(dTD, (long long)ND * D);
  gcnhet_scatter_add<<<cdivll((long long)E_td * 32, TB), TB, 0, stream>>>(
      tMSG, src_td, dst_td, ew_td, dTD, E_td);

  // ---- doc = relu(wd/deg + td/deg); pooled = segment_max; head
  gcnhet_doc_combine<<<cdivll((long long)ND * D, TB), TB, 0, stream>>>(
      dWD, dTD, deg_wd, deg_td, (long long)ND * D);
  ZERO(pooled, 16 * D);
  gcnhet_pool_max<<<cdivll((long long)ND * D, TB), TB, 0, stream>>>(
      dWD, doc_gid, (unsigned*)pooled, (long long)ND * D);
  gcnhet_final<<<1, 512, 0, stream>>>(pooled, outW, outb, y_data, (float*)d_out);
}
